// DeductionNetworkSingleLayer_66992899883454
// MI455X (gfx1250) — matrix operand order repaired and checked by reading, not compiled, not run on hardware
//
#include <hip/hip_runtime.h>

// ---------------------------------------------------------------------------
// Types for CDNA5 (gfx1250) wave32 WMMA
// ---------------------------------------------------------------------------
typedef __bf16 bf16_t;
typedef __attribute__((ext_vector_type(16))) __bf16 v16bf;
typedef __attribute__((ext_vector_type(8)))  __bf16 v8bf;
typedef __attribute__((ext_vector_type(8)))  float  v8f;

#define WMMA_BF16(a, b, c) \
  __builtin_amdgcn_wmma_f32_16x16x32_bf16(false, (a), false, (b), (short)0, (c), false, false)

// Problem constants (from the reference)
#define BATCH 2
#define SQL   2048
#define SKL   2048
#define EDIM  256
#define SDIM  256
#define NHEAD 8
#define HIDD  512

// ---------------------------------------------------------------------------
// Fragment loaders (wave32; contiguous k order, the same for A and B)
// A (16x32 bf16, row-major source, ld in elements):
//   lane: m = lane&15, half = lane>>4; elem i -> k = 16*half + i
//   => one contiguous 32B run at k = 16*half
// B (32x16 bf16, stored [n][k] contiguous-k):
//   lane: n = nb + (lane&15), half = lane>>4; elem i -> k = 16*half + i
//   => one contiguous 32B run
// ---------------------------------------------------------------------------
__device__ __forceinline__ v16bf a_frag_rowmajor(const bf16_t* tile, int ld, int kbase) {
  const int lane = threadIdx.x & 31;
  const int m = lane & 15, hgrp = lane >> 4;
  const bf16_t* p = tile + (size_t)m * ld + kbase + 16 * hgrp;
  v16bf a;
  *((v8bf*)&a)     = *((const v8bf*)p);
  *((v8bf*)&a + 1) = *((const v8bf*)(p + 8));
  return a;
}

__device__ __forceinline__ v16bf b_frag_nk(const bf16_t* tile, int ld, int nb, int kbase) {
  const int lane = threadIdx.x & 31;
  const int n = nb + (lane & 15), hgrp = lane >> 4;
  return *((const v16bf*)(tile + (size_t)n * ld + kbase + 16 * hgrp));
}

// ---------------------------------------------------------------------------
// Generic WMMA GEMM:  C[M,N] = A[M,K] @ Bw[N,K]^T + bias   (torch Linear)
// Compile-time specialization (no runtime branches in hot code):
//   AF32:  A operand is f32 (else bf16)
//   MODE:  0: outf[row*N+col] f32       1: outb[row*N+col] bf16
//          2: q/k head split  outb[((b*NH+h)*SQL + r)*EDIM + e]
//          3: v head split^T  outb[((b*NH+h)*SDIM + d)*SKL + r]
//   RELU:  apply max(0,.)
// 64x64 block tile, 4 waves (128 thr), K-step 32.
// ---------------------------------------------------------------------------
template <bool AF32, int MODE, bool RELU>
__global__ __launch_bounds__(128)
void gemm_kernel(const void* __restrict__ Asrc,
                 const float* __restrict__ Bw, const float* __restrict__ bias,
                 float* __restrict__ outf, bf16_t* __restrict__ outb,
                 int M, int N, int K)
{
  __shared__ __align__(64) bf16_t As[64][32];
  __shared__ __align__(64) bf16_t Bs[64][32];

  const int tid  = threadIdx.x;
  const int wave = tid >> 5;
  const int m0   = blockIdx.y * 64;
  const int n0   = blockIdx.x * 64;

  v8f acc[4] = {};

  for (int k0 = 0; k0 < K; k0 += 32) {
    // Cooperative stage: 64x32 of A and B, 16 contiguous elements per thread.
    const int base = tid * 16;
    const int mr = base >> 5;       // 0..63
    const int kk = base & 31;       // 0 or 16
    if (AF32) {
      const float* src = (const float*)Asrc + (size_t)(m0 + mr) * K + k0 + kk;
#pragma unroll
      for (int i = 0; i < 16; ++i) As[mr][kk + i] = (bf16_t)src[i];
    } else {
      const bf16_t* src = (const bf16_t*)Asrc + (size_t)(m0 + mr) * K + k0 + kk;
#pragma unroll
      for (int i = 0; i < 16; ++i) As[mr][kk + i] = src[i];
    }
    {
      const float* src = Bw + (size_t)(n0 + mr) * K + k0 + kk;
#pragma unroll
      for (int i = 0; i < 16; ++i) Bs[mr][kk + i] = (bf16_t)src[i];
    }
    __syncthreads();

    // Batch all fragment loads first, then the 4 WMMAs: the DS loads issue as
    // one clause and overlap with matrix ops instead of wait-per-WMMA.
    v16bf a = a_frag_rowmajor(&As[wave * 16][0], 32, 0);
    v16bf bfr[4];
#pragma unroll
    for (int nt = 0; nt < 4; ++nt) bfr[nt] = b_frag_nk(&Bs[0][0], 32, nt * 16, 0);
#pragma unroll
    for (int nt = 0; nt < 4; ++nt) acc[nt] = WMMA_BF16(a, bfr[nt], acc[nt]);
    __syncthreads();
  }

  // Epilogue: C layout row = r + 8*half, col = lane&15 (§7.12.2)
  const int lane = tid & 31;
  const int nlo  = lane & 15, hgrp = lane >> 4;
#pragma unroll
  for (int nt = 0; nt < 4; ++nt) {
    const int col = n0 + nt * 16 + nlo;
    const float bvv = bias[col];
#pragma unroll
    for (int r = 0; r < 8; ++r) {
      const int row = m0 + wave * 16 + r + 8 * hgrp;
      float v = acc[nt][r] + bvv;
      if (RELU) v = v > 0.0f ? v : 0.0f;
      if (MODE == 2) {
        // rows-per-batch = SQL (power of 2), head width = EDIM (power of 2)
        const int bb = row >> 11, rr = row & (SQL - 1);
        const int hh = col >> 8,  ee = col & (EDIM - 1);
        outb[(((size_t)(bb * NHEAD + hh) * SQL + rr) * EDIM + ee)] = (bf16_t)v;
      } else if (MODE == 3) {
        const int bb = row >> 11, rr = row & (SKL - 1);
        const int hh = col >> 8,  dd = col & (SDIM - 1);
        outb[(((size_t)(bb * NHEAD + hh) * SDIM + dd) * SKL + rr)] = (bf16_t)v;
      } else if (MODE == 1) {
        outb[(size_t)row * N + col] = (bf16_t)v;
      } else {
        outf[(size_t)row * N + col] = v;
      }
    }
  }
}

// ---------------------------------------------------------------------------
// Flash attention:  O = softmax(scale * q @ k^T) @ v
// q: [B*Hh, SQ, 256] bf16 ; k: [B*Hh, SK, 256] bf16 ; vt: [B*Hh, 256, SK] bf16
// Out row = b*SQ + q, col = h*256 + d, row stride ldo.
//   OUTB:   write bf16 (else f32)
//   LSCALE: multiply by learned scalar *scale_ptr (else scale_ptr unused)
// 4 waves/WG, each wave owns a 16-row q tile; 32-key K/V tiles staged in LDS.
// q A-fragments are loop-invariant and held in VGPRs for all 64 key tiles.
// ---------------------------------------------------------------------------
template <bool OUTB, bool LSCALE>
__global__ __launch_bounds__(128)
void flash_kernel(const bf16_t* __restrict__ qg, const bf16_t* __restrict__ kg,
                  const bf16_t* __restrict__ vtg,
                  float* __restrict__ outf, bf16_t* __restrict__ outb,
                  const float* __restrict__ scale_ptr, float scale_mul,
                  int SQn, int SKn, int Hh, int ldo)
{
  __shared__ __align__(64) bf16_t Ks[32][256];      // [key][e]      16 KB
  __shared__ __align__(64) bf16_t Vs[256][32];      // [d][key]      16 KB
  __shared__ __align__(64) bf16_t Ps[4][16][32];    // per-wave P     4 KB

  const int bh   = blockIdx.x;
  const int b    = bh / Hh, h = bh % Hh;
  const int tid  = threadIdx.x;
  const int wave = tid >> 5;
  const int lane = tid & 31;
  const int nlo  = lane & 15, hgrp = lane >> 4;
  const int qloc = blockIdx.y * 64 + wave * 16;     // wave's q-tile start (within head)

  const bf16_t* qbase  = qg  + (size_t)bh * SQn * 256;
  const bf16_t* kbase  = kg  + (size_t)bh * SKn * 256;
  const bf16_t* vtbase = vtg + (size_t)bh * 256 * SKn;
  const float scale = LSCALE ? (scale_mul * scale_ptr[0]) : scale_mul;

  // q fragments: loop-invariant, keep resident (8 x 8 VGPRs)
  v16bf af[8];
#pragma unroll
  for (int eb = 0; eb < 8; ++eb)
    af[eb] = a_frag_rowmajor(qbase + (size_t)qloc * 256, 256, eb * 32);

  float mrun[8], lrun[8];
  v8f o[16] = {};
#pragma unroll
  for (int r = 0; r < 8; ++r) { mrun[r] = -1e30f; lrun[r] = 0.0f; }

  const int nkt = SKn / 32;
  for (int kt = 0; kt < nkt; ++kt) {
    // ---- cooperative LDS staging: 32 keys of K (row-major e) and V^T ----
    {
      const int krow = tid >> 2, seg = tid & 3;     // 32 rows x 4 segs of 64
      const bf16_t* src = kbase + (size_t)(kt * 32 + krow) * 256 + seg * 64;
      v8bf* dst = (v8bf*)&Ks[krow][seg * 64];
      const v8bf* s8 = (const v8bf*)src;
#pragma unroll
      for (int i = 0; i < 8; ++i) dst[i] = s8[i];
      if (kt + 1 < nkt)
        __builtin_prefetch(kbase + (size_t)((kt + 1) * 32 + krow) * 256 + seg * 64, 0, 0);

      const int d0 = tid * 2;                       // 2 d-rows per thread
#pragma unroll
      for (int rr = 0; rr < 2; ++rr) {
        const bf16_t* vsrc = vtbase + (size_t)(d0 + rr) * SKn + kt * 32;
        v8bf* vdst = (v8bf*)&Vs[d0 + rr][0];
        const v8bf* vs8 = (const v8bf*)vsrc;
#pragma unroll
        for (int i = 0; i < 4; ++i) vdst[i] = vs8[i];
        if (kt + 1 < nkt)
          __builtin_prefetch(vtbase + (size_t)(d0 + rr) * SKn + (kt + 1) * 32, 0, 0);
      }
    }
    __syncthreads();

    // ---- scores: two 16x16 tiles over this wave's 16 q rows (K = 256) ----
    // Batch 4 B-fragment loads per 4 WMMAs so DS latency overlaps matrix ops.
    v8f s0 = {}, s1 = {};
#pragma unroll
    for (int eb = 0; eb < 8; eb += 2) {
      v16bf kb[4];
      kb[0] = b_frag_nk(&Ks[0][0], 256,  0, eb * 32);
      kb[1] = b_frag_nk(&Ks[0][0], 256, 16, eb * 32);
      kb[2] = b_frag_nk(&Ks[0][0], 256,  0, (eb + 1) * 32);
      kb[3] = b_frag_nk(&Ks[0][0], 256, 16, (eb + 1) * 32);
      s0 = WMMA_BF16(af[eb],     kb[0], s0);
      s1 = WMMA_BF16(af[eb],     kb[1], s1);
      s0 = WMMA_BF16(af[eb + 1], kb[2], s0);
      s1 = WMMA_BF16(af[eb + 1], kb[3], s1);
    }

    // ---- online softmax (C layout: row r+8*hgrp, col nlo / 16+nlo) ----
    float alph[8];
#pragma unroll
    for (int r = 0; r < 8; ++r) {
      float v0 = s0[r] * scale, v1 = s1[r] * scale;
      float mx = fmaxf(v0, v1);
#pragma unroll
      for (int off = 1; off < 16; off <<= 1) mx = fmaxf(mx, __shfl_xor(mx, off, 32));
      const float mnew = fmaxf(mrun[r], mx);
      alph[r] = __expf(mrun[r] - mnew);
      const float p0 = __expf(v0 - mnew);
      const float p1 = __expf(v1 - mnew);
      float ps = p0 + p1;
#pragma unroll
      for (int off = 1; off < 16; off <<= 1) ps += __shfl_xor(ps, off, 32);
      lrun[r] = lrun[r] * alph[r] + ps;
      mrun[r] = mnew;
      Ps[wave][r + 8 * hgrp][nlo]      = (bf16_t)p0;
      Ps[wave][r + 8 * hgrp][16 + nlo] = (bf16_t)p1;
    }
#pragma unroll
    for (int dt = 0; dt < 16; ++dt)
#pragma unroll
      for (int r = 0; r < 8; ++r) o[dt][r] *= alph[r];

    // same-wave DS ops are in-order; wait for P writes before fragment reads
    asm volatile("s_wait_dscnt 0" ::: "memory");

    // ---- O += P(16x32) @ V^T  across all 256 value dims ----
    v16bf pf = a_frag_rowmajor(&Ps[wave][0][0], 32, 0);
#pragma unroll
    for (int dt = 0; dt < 16; dt += 4) {
      v16bf bv[4];
#pragma unroll
      for (int j = 0; j < 4; ++j) bv[j] = b_frag_nk(&Vs[0][0], 32, (dt + j) * 16, 0);
#pragma unroll
      for (int j = 0; j < 4; ++j) o[dt + j] = WMMA_BF16(pf, bv[j], o[dt + j]);
    }
    __syncthreads();
  }

  // ---- epilogue: O / l ----
  float inv[8];
#pragma unroll
  for (int r = 0; r < 8; ++r) inv[r] = 1.0f / lrun[r];
#pragma unroll
  for (int dt = 0; dt < 16; ++dt) {
    const int col = h * 256 + dt * 16 + nlo;
#pragma unroll
    for (int r = 0; r < 8; ++r) {
      const size_t row = (size_t)b * SQn + qloc + r + 8 * hgrp;
      const float v = o[dt][r] * inv[r];
      if (OUTB) outb[row * ldo + col] = (bf16_t)v;
      else      outf[row * ldo + col] = v;
    }
  }
}

// ---------------------------------------------------------------------------
// Elementwise helpers
// ---------------------------------------------------------------------------
__global__ void cvt_bf16_kernel(const float* __restrict__ in, bf16_t* __restrict__ out, size_t n) {
  size_t i = (size_t)blockIdx.x * blockDim.x + threadIdx.x;
  if (i < n) out[i] = (bf16_t)in[i];
}

// A [B,SK,S] f32 -> At [B,S,SK] bf16
__global__ void transpose_cvt_kernel(const float* __restrict__ in, bf16_t* __restrict__ out) {
  size_t i = (size_t)blockIdx.x * blockDim.x + threadIdx.x;
  const size_t total = (size_t)BATCH * SKL * SDIM;
  if (i >= total) return;
  const int s  = (int)(i & (SDIM - 1));
  const size_t t = i >> 8;
  const int sk = (int)(t & (SKL - 1));
  const int bb = (int)(t >> 11);
  out[((size_t)bb * SDIM + s) * SKL + sk] = (bf16_t)in[i];
}

// out = LayerNorm(x + y) ; one 256-thread block per row, C = 256
__global__ __launch_bounds__(256)
void add_ln_kernel(const float* __restrict__ x, const float* __restrict__ y,
                   const float* __restrict__ g, const float* __restrict__ bta,
                   float* __restrict__ out)
{
  __shared__ float red[256];
  const int row = blockIdx.x, c = threadIdx.x;
  const size_t idx = (size_t)row * 256 + c;
  const float v = x[idx] + y[idx];
  red[c] = v; __syncthreads();
  for (int s = 128; s > 0; s >>= 1) { if (c < s) red[c] += red[c + s]; __syncthreads(); }
  const float mean = red[0] * (1.0f / 256.0f);
  __syncthreads();
  const float d = v - mean;
  red[c] = d * d; __syncthreads();
  for (int s = 128; s > 0; s >>= 1) { if (c < s) red[c] += red[c + s]; __syncthreads(); }
  const float var = red[0] * (1.0f / 256.0f);
  out[idx] = d * rsqrtf(var + 1e-5f) * g[c] + bta[c];
}

// ---------------------------------------------------------------------------
// Orchestration
// ---------------------------------------------------------------------------
extern "C" void kernel_launch(void* const* d_in, const int* in_sizes, int n_in,
                              void* d_out, int out_size, void* d_ws, size_t ws_size,
                              hipStream_t stream) {
  (void)in_sizes; (void)n_in; (void)out_size; (void)ws_size;
  const float* Q   = (const float*)d_in[0];
  const float* H   = (const float*)d_in[1];
  const float* A   = (const float*)d_in[2];
  const float* wq  = (const float*)d_in[3];  const float* bq = (const float*)d_in[4];
  const float* wk  = (const float*)d_in[5];  const float* bk = (const float*)d_in[6];
  const float* wv  = (const float*)d_in[7];  const float* bv = (const float*)d_in[8];
  const float* wo  = (const float*)d_in[9];  const float* bo = (const float*)d_in[10];
  const float* attn_scale = (const float*)d_in[11];
  const float* ln_g = (const float*)d_in[12]; const float* ln_b = (const float*)d_in[13];
  const float* w1  = (const float*)d_in[14]; const float* b1 = (const float*)d_in[15];
  const float* w2  = (const float*)d_in[16]; const float* b2 = (const float*)d_in[17];

  // Workspace layout (bytes)
  char* ws = (char*)d_ws;
  const size_t SZ_HEADS = (size_t)BATCH * NHEAD * SQL * EDIM * 2;   // 16.78 MB
  bf16_t* qbf  = (bf16_t*)(ws);                       // [B,NH,SQ,E]   (reused: Qbf)
  bf16_t* kbf  = (bf16_t*)(ws + SZ_HEADS);            // [B,NH,SK,E]   (reused: Hbf)
  bf16_t* vtbf = (bf16_t*)(ws + 2 * SZ_HEADS);        // [B,NH,S,SK]   (reused: Atbf)
  bf16_t* ctx  = (bf16_t*)(ws + 3 * SZ_HEADS);        // [B,SQ,NH*S]
  char* p = ws + 4 * SZ_HEADS;
  float* Am   = (float*)p;                 p += (size_t)BATCH * SQL * SDIM * 4;
  float* attn = (float*)p;                 p += (size_t)BATCH * SQL * SDIM * 4;
  float* Ad   = (float*)p;                 p += (size_t)BATCH * SQL * SDIM * 4;
  bf16_t* ff1 = (bf16_t*)p;                p += (size_t)BATCH * SQL * HIDD * 2;
  float* ff2  = (float*)p;

  const int M = BATCH * SQL;               // 4096
  const dim3 blk128(128), blk256(256);

  // 1-3) q/k/v projections -> bf16 head-split layouts
  gemm_kernel<true, 2, false><<<dim3((NHEAD*EDIM)/64, M/64), blk128, 0, stream>>>(
      Q, wq, bq, nullptr, qbf, M, NHEAD*EDIM, EDIM);
  gemm_kernel<true, 2, false><<<dim3((NHEAD*EDIM)/64, M/64), blk128, 0, stream>>>(
      H, wk, bk, nullptr, kbf, M, NHEAD*EDIM, EDIM);
  gemm_kernel<true, 3, false><<<dim3((NHEAD*SDIM)/64, M/64), blk128, 0, stream>>>(
      A, wv, bv, nullptr, vtbf, M, NHEAD*SDIM, SDIM);

  // 4) MHA flash attention -> ctx bf16 [B,SQ,NH*S]
  flash_kernel<true, false><<<dim3(BATCH*NHEAD, SQL/64), blk128, 0, stream>>>(
      qbf, kbf, vtbf, nullptr, ctx, nullptr, 0.0625f /* 1/sqrt(256) */,
      SQL, SKL, NHEAD, NHEAD*SDIM);

  // 5) output projection: Am = ctx @ wo^T + bo
  gemm_kernel<false, 0, false><<<dim3(SDIM/64, M/64), blk128, 0, stream>>>(
      ctx, wo, bo, Am, nullptr, M, SDIM, NHEAD*SDIM);

  // 6) bf16 copies for branch 2 (reuse q/k/v regions, all dead now)
  {
    const size_t n = (size_t)BATCH * SQL * EDIM;      // 1M
    cvt_bf16_kernel<<<dim3((unsigned)((n + 255) / 256)), blk256, 0, stream>>>(Q, qbf, n);
    cvt_bf16_kernel<<<dim3((unsigned)((n + 255) / 256)), blk256, 0, stream>>>(H, kbf, n);
    transpose_cvt_kernel<<<dim3((unsigned)((n + 255) / 256)), blk256, 0, stream>>>(A, vtbf);
  }

  // 7) Attension branch: attn = softmax(Q@H^T * attn_scale) @ A   (f32 out)
  flash_kernel<false, true><<<dim3(BATCH, SQL/64), blk128, 0, stream>>>(
      qbf, kbf, vtbf, attn, nullptr, attn_scale, 1.0f, SQL, SKL, 1, SDIM);

  // 8) Ad = LN(Am + attn)
  add_ln_kernel<<<dim3(M), blk256, 0, stream>>>(Am, attn, ln_g, ln_b, Ad);

  // 9) ff1 = relu(Ad @ w1^T + b1)  (bf16)
  gemm_kernel<true, 1, true><<<dim3(HIDD/64, M/64), blk128, 0, stream>>>(
      Ad, w1, b1, nullptr, ff1, M, HIDD, SDIM);

  // 10) ff2 = ff1 @ w2^T + b2  (f32)
  gemm_kernel<false, 0, false><<<dim3(SDIM/64, M/64), blk128, 0, stream>>>(
      ff1, w2, b2, ff2, nullptr, M, SDIM, HIDD);

  // 11) out = LN(ff2 + Ad)
  add_ln_kernel<<<dim3(M), blk256, 0, stream>>>(ff2, Ad, ln_g, ln_b, (float*)d_out);
}
